// Reconstruction_loss_10127532884154
// MI455X (gfx1250) — compile-verified, matrix operand order checked by reading, not run on hardware
//
#include <hip/hip_runtime.h>

#define IN_DIM 128
#define HID_DIM 256
#define OUT_DIM 128

typedef __attribute__((ext_vector_type(16))) __bf16 v16bf;
typedef __attribute__((ext_vector_type(8)))  __bf16 v8bf;
typedef __attribute__((ext_vector_type(8)))  float  v8f;

__device__ __forceinline__ __bf16 f2bf(float f) {
    union { float f; unsigned u; } x; x.f = f;
    unsigned r = x.u + 0x7FFFu + ((x.u >> 16) & 1u);   // round-to-nearest-even
    union { unsigned short s; __bf16 b; } y;
    y.s = (unsigned short)(r >> 16);
    return y.b;
}

// load 16 bf16 for one WMMA fragment: two contiguous 16B chunks (K-octets per ISA layout)
__device__ __forceinline__ v16bf ldfrag(const __bf16* __restrict__ p, int ka) {
    v8bf lo = *(const v8bf*)(p + ka);
    v8bf hi = *(const v8bf*)(p + ka + 16);
    return __builtin_shufflevector(lo, hi, 0, 1, 2, 3, 4, 5, 6, 7,
                                          8, 9, 10, 11, 12, 13, 14, 15);
}

// ---------------- degree / norm ----------------
__global__ void deg_init_kernel(float* deg, int N) {
    int i = blockIdx.x * blockDim.x + threadIdx.x;
    if (i < N) deg[i] = 1.0f;  // self-loop
}

__global__ void deg_count_kernel(const int* __restrict__ dst, float* deg, int E) {
    int e = blockIdx.x * blockDim.x + threadIdx.x;
    if (e < E) atomicAdd(&deg[dst[e]], 1.0f);
}

__global__ void make_dinv_kernel(float* deg, int N) {
    int i = blockIdx.x * blockDim.x + threadIdx.x;
    if (i < N) {
        float d = deg[i];
        deg[i] = (d > 0.0f) ? rsqrtf(d) : 0.0f;
    }
}

// ---------------- conversions ----------------
__global__ void f32_to_bf16_kernel(const float* __restrict__ in, __bf16* __restrict__ out, long n) {
    long i = (long)blockIdx.x * blockDim.x + threadIdx.x;
    if (i < n) out[i] = f2bf(in[i]);
}

// W [K x Nc] f32 row-major -> Wt [Nc x K] bf16 row-major
__global__ void transpose_bf16_kernel(const float* __restrict__ W, __bf16* __restrict__ Wt,
                                      int K, int Nc) {
    long i = (long)blockIdx.x * blockDim.x + threadIdx.x;
    if (i < (long)K * Nc) {
        int k = (int)(i / Nc);
        int n = (int)(i % Nc);
        Wt[(long)n * K + k] = f2bf(W[i]);
    }
}

// ---------------- WMMA GEMM: C[MxNc] = A[MxK] * Bt^T ----------------
// A  : [M x K]  bf16 row-major
// Bt : [Nc x K] bf16 row-major (B[k][n] = Bt[n*K+k])
// One wave computes a 64x64 C block (4x4 tiles of 16x16): 16 WMMAs per 8 fragment
// loads -> 32 bf16-FLOP per loaded byte; operands are L2-resident.
template <int K, bool NT>
__global__ __launch_bounds__(256) void gemm_bf16_kernel(
        const __bf16* __restrict__ A, const __bf16* __restrict__ Bt,
        float* __restrict__ C, int M, int Nc) {
    const int wavesPerBlock = blockDim.x >> 5;
    const int wave = threadIdx.x >> 5;
    const int lane = threadIdx.x & 31;
    const int bm = (M + 63) >> 6;
    const int bn = (Nc + 63) >> 6;
    const long blk = (long)blockIdx.x * wavesPerBlock + wave;
    if (blk >= (long)bm * bn) return;            // wave-uniform: EXEC all-ones for WMMA
    const int ibm = (int)(blk / bn);
    const int ibn = (int)(blk % bn);

    const int half = lane >> 4;                  // K-octet group select per ISA 16-bit layout
    const int l    = lane & 15;

    const __bf16* __restrict__ ap[4];
    const __bf16* __restrict__ bp[4];
#pragma unroll
    for (int i = 0; i < 4; ++i) {
        int r = ibm * 64 + i * 16 + l;  if (r > M - 1)  r = M - 1;
        int c = ibn * 64 + i * 16 + l;  if (c > Nc - 1) c = Nc - 1;
        ap[i] = A  + (long)r * K;
        bp[i] = Bt + (long)c * K;
    }

    v8f acc[4][4];
#pragma unroll
    for (int i = 0; i < 4; ++i)
#pragma unroll
        for (int j = 0; j < 4; ++j)
            acc[i][j] = (v8f){0.f,0.f,0.f,0.f,0.f,0.f,0.f,0.f};

#pragma unroll
    for (int k0 = 0; k0 < K; k0 += 32) {
        const int ka = k0 + half * 8;    // lanes 0-15: K 0..7/16..23; lanes 16-31: K 8..15/24..31
        v16bf a0 = ldfrag(ap[0], ka);
        v16bf a1 = ldfrag(ap[1], ka);
        v16bf a2 = ldfrag(ap[2], ka);
        v16bf a3 = ldfrag(ap[3], ka);
        // B fragments consumed in pairs to bound live registers
        {
            v16bf b0 = ldfrag(bp[0], ka);
            v16bf b1 = ldfrag(bp[1], ka);
            acc[0][0] = __builtin_amdgcn_wmma_f32_16x16x32_bf16(false, a0, false, b0, (short)0, acc[0][0], false, false);
            acc[1][0] = __builtin_amdgcn_wmma_f32_16x16x32_bf16(false, a1, false, b0, (short)0, acc[1][0], false, false);
            acc[2][0] = __builtin_amdgcn_wmma_f32_16x16x32_bf16(false, a2, false, b0, (short)0, acc[2][0], false, false);
            acc[3][0] = __builtin_amdgcn_wmma_f32_16x16x32_bf16(false, a3, false, b0, (short)0, acc[3][0], false, false);
            acc[0][1] = __builtin_amdgcn_wmma_f32_16x16x32_bf16(false, a0, false, b1, (short)0, acc[0][1], false, false);
            acc[1][1] = __builtin_amdgcn_wmma_f32_16x16x32_bf16(false, a1, false, b1, (short)0, acc[1][1], false, false);
            acc[2][1] = __builtin_amdgcn_wmma_f32_16x16x32_bf16(false, a2, false, b1, (short)0, acc[2][1], false, false);
            acc[3][1] = __builtin_amdgcn_wmma_f32_16x16x32_bf16(false, a3, false, b1, (short)0, acc[3][1], false, false);
        }
        {
            v16bf b2 = ldfrag(bp[2], ka);
            v16bf b3 = ldfrag(bp[3], ka);
            acc[0][2] = __builtin_amdgcn_wmma_f32_16x16x32_bf16(false, a0, false, b2, (short)0, acc[0][2], false, false);
            acc[1][2] = __builtin_amdgcn_wmma_f32_16x16x32_bf16(false, a1, false, b2, (short)0, acc[1][2], false, false);
            acc[2][2] = __builtin_amdgcn_wmma_f32_16x16x32_bf16(false, a2, false, b2, (short)0, acc[2][2], false, false);
            acc[3][2] = __builtin_amdgcn_wmma_f32_16x16x32_bf16(false, a3, false, b2, (short)0, acc[3][2], false, false);
            acc[0][3] = __builtin_amdgcn_wmma_f32_16x16x32_bf16(false, a0, false, b3, (short)0, acc[0][3], false, false);
            acc[1][3] = __builtin_amdgcn_wmma_f32_16x16x32_bf16(false, a1, false, b3, (short)0, acc[1][3], false, false);
            acc[2][3] = __builtin_amdgcn_wmma_f32_16x16x32_bf16(false, a2, false, b3, (short)0, acc[2][3], false, false);
            acc[3][3] = __builtin_amdgcn_wmma_f32_16x16x32_bf16(false, a3, false, b3, (short)0, acc[3][3], false, false);
        }
    }

    // C store per 16x16 f32 layout: VGPR r -> row r + 8*half; col = lane&15
#pragma unroll
    for (int i = 0; i < 4; ++i) {
#pragma unroll
        for (int j = 0; j < 4; ++j) {
            const int colj = ibn * 64 + j * 16 + l;
            if (colj < Nc) {
#pragma unroll
                for (int r = 0; r < 8; ++r) {
                    const int row = ibm * 64 + i * 16 + r + 8 * half;
                    if (row < M) {
                        if (NT) __builtin_nontemporal_store(acc[i][j][r], &C[(long)row * Nc + colj]);
                        else    C[(long)row * Nc + colj] = acc[i][j][r];
                    }
                }
            }
        }
    }
}

// ---------------- edge aggregation: agg[dst] += xw[src] * dinv[src]*dinv[dst] ----------------
__global__ void edge_agg_kernel(const float* __restrict__ xw,
                                const int* __restrict__ src, const int* __restrict__ dst,
                                const float* __restrict__ dinv,
                                float* __restrict__ agg, int E, int F) {
    long idx = (long)blockIdx.x * blockDim.x + threadIdx.x;
    if (idx >= (long)E * F) return;
    int e = (int)(idx / F);
    int f = (int)(idx % F);
    int s = src[e], d = dst[e];
    float norm = dinv[s] * dinv[d];
    atomicAdd(&agg[(long)d * F + f], xw[(long)s * F + f] * norm);
}

// ---------------- finalize: v = act(agg + xw*dinv^2 + bias) ----------------
__global__ void finalize_kernel(const float* __restrict__ agg, const float* __restrict__ xw,
                                const float* __restrict__ dinv, const float* __restrict__ bias,
                                float* __restrict__ outf, __bf16* __restrict__ outb,
                                int N, int F, int relu) {
    long idx = (long)blockIdx.x * blockDim.x + threadIdx.x;
    if (idx >= (long)N * F) return;
    int n = (int)(idx / F);
    int f = (int)(idx % F);
    float di = dinv[n];
    float v = agg[idx] + xw[idx] * di * di + bias[f];
    if (relu) v = fmaxf(v, 0.0f);
    if (outf) outf[idx] = v;
    if (outb) outb[idx] = f2bf(v);
}

// ---------------- host helpers ----------------
template <int K, bool NT>
static void launch_gemm(const __bf16* A, const __bf16* Bt, float* C,
                        int M, int Nc, hipStream_t stream) {
    long nBlk = (long)((M + 63) / 64) * ((Nc + 63) / 64);
    const int wavesPerBlock = 8;                 // 256 threads = 8 wave32
    long blocks = (nBlk + wavesPerBlock - 1) / wavesPerBlock;
    gemm_bf16_kernel<K, NT><<<dim3((unsigned)blocks), dim3(32 * wavesPerBlock), 0, stream>>>(
        A, Bt, C, M, Nc);
}

extern "C" void kernel_launch(void* const* d_in, const int* in_sizes, int n_in,
                              void* d_out, int out_size, void* d_ws, size_t ws_size,
                              hipStream_t stream) {
    const float* h   = (const float*)d_in[0];
    const float* W1a = (const float*)d_in[1];
    const float* b1a = (const float*)d_in[2];
    const float* W2a = (const float*)d_in[3];
    const float* b2a = (const float*)d_in[4];
    const float* W1s = (const float*)d_in[5];
    const float* b1s = (const float*)d_in[6];
    const float* W2s = (const float*)d_in[7];
    const float* b2s = (const float*)d_in[8];
    const int*   ei  = (const int*)d_in[9];

    const int N = in_sizes[0] / IN_DIM;
    const int E = in_sizes[9] / 2;
    const int* src = ei;          // edge_index row 0
    const int* dst = ei + E;      // edge_index row 1

    // ---- workspace layout ----
    char* wsb = (char*)d_ws;
    float* dinv = (float*)wsb;                       wsb += (size_t)N * sizeof(float);
    float* xw   = (float*)wsb;                       wsb += (size_t)N * HID_DIM * sizeof(float);
    float* agg  = (float*)wsb;                       wsb += (size_t)N * HID_DIM * sizeof(float);
    __bf16* hb  = (__bf16*)wsb;                      wsb += (size_t)N * IN_DIM  * sizeof(__bf16);
    __bf16* h1b = (__bf16*)wsb;                      wsb += (size_t)N * HID_DIM * sizeof(__bf16);
    __bf16* hsb = (__bf16*)wsb;                      wsb += (size_t)N * OUT_DIM * sizeof(__bf16);
    __bf16* w1t = (__bf16*)wsb;                      wsb += (size_t)IN_DIM * HID_DIM * sizeof(__bf16);
    __bf16* w2t = (__bf16*)wsb;                      wsb += (size_t)HID_DIM * OUT_DIM * sizeof(__bf16);

    // ---- output layout: x_ [N*OUT] | s_ [N*N] | h [N*IN] ----
    float* out   = (float*)d_out;
    float* x_out = out;
    float* s_out = out + (size_t)N * OUT_DIM;
    float* h_out = s_out + (size_t)N * N;

    // ---- symmetric normalization ----
    deg_init_kernel<<<(N + 255) / 256, 256, 0, stream>>>(dinv, N);
    deg_count_kernel<<<(E + 255) / 256, 256, 0, stream>>>(dst, dinv, E);
    make_dinv_kernel<<<(N + 255) / 256, 256, 0, stream>>>(dinv, N);

    // ---- h -> bf16 once ----
    {
        long n = (long)N * IN_DIM;
        f32_to_bf16_kernel<<<(unsigned)((n + 255) / 256), 256, 0, stream>>>(h, hb, n);
    }

    // ---- two GCN branches ----
    for (int br = 0; br < 2; ++br) {
        const float* W1 = br ? W1s : W1a;
        const float* B1 = br ? b1s : b1a;
        const float* W2 = br ? W2s : W2a;
        const float* B2 = br ? b2s : b2a;

        // weights -> transposed bf16: W1 [IN x HID] -> w1t [HID x IN], W2 [HID x OUT] -> w2t [OUT x HID]
        {
            long n1 = (long)IN_DIM * HID_DIM;
            transpose_bf16_kernel<<<(unsigned)((n1 + 255) / 256), 256, 0, stream>>>(W1, w1t, IN_DIM, HID_DIM);
            long n2 = (long)HID_DIM * OUT_DIM;
            transpose_bf16_kernel<<<(unsigned)((n2 + 255) / 256), 256, 0, stream>>>(W2, w2t, HID_DIM, OUT_DIM);
        }

        // layer 1: xw = h @ W1 ; aggregate ; h1b = bf16(relu(...))
        launch_gemm<IN_DIM, false>(hb, w1t, xw, N, HID_DIM, stream);
        hipMemsetAsync(agg, 0, (size_t)N * HID_DIM * sizeof(float), stream);
        {
            long tot = (long)E * HID_DIM;
            edge_agg_kernel<<<(unsigned)((tot + 255) / 256), 256, 0, stream>>>(
                xw, src, dst, dinv, agg, E, HID_DIM);
            long tn = (long)N * HID_DIM;
            finalize_kernel<<<(unsigned)((tn + 255) / 256), 256, 0, stream>>>(
                agg, xw, dinv, B1, (float*)nullptr, h1b, N, HID_DIM, /*relu=*/1);
        }

        // layer 2: xw = h1 @ W2 ; aggregate ; branch outputs
        launch_gemm<HID_DIM, false>(h1b, w2t, xw, N, OUT_DIM, stream);
        hipMemsetAsync(agg, 0, (size_t)N * OUT_DIM * sizeof(float), stream);
        {
            long tot = (long)E * OUT_DIM;
            edge_agg_kernel<<<(unsigned)((tot + 255) / 256), 256, 0, stream>>>(
                xw, src, dst, dinv, agg, E, OUT_DIM);
            long tn = (long)N * OUT_DIM;
            if (br == 0) {
                // attribute branch: x_ output in f32
                finalize_kernel<<<(unsigned)((tn + 255) / 256), 256, 0, stream>>>(
                    agg, xw, dinv, B2, x_out, (__bf16*)nullptr, N, OUT_DIM, /*relu=*/0);
            } else {
                // structure branch: only needed as bf16 GEMM operand for s_
                finalize_kernel<<<(unsigned)((tn + 255) / 256), 256, 0, stream>>>(
                    agg, xw, dinv, B2, (float*)nullptr, hsb, N, OUT_DIM, /*relu=*/0);
            }
        }
    }

    // ---- s_ = hs @ hs^T (dominant GEMM: 10000x10000x128) ----
    // NT stores: 400 MB of s_ is written once and never re-read on device.
    launch_gemm<OUT_DIM, true>(hsb, hsb, s_out, N, N, stream);

    // ---- third output is the input h itself ----
    hipMemcpyAsync(h_out, h, (size_t)N * IN_DIM * sizeof(float),
                   hipMemcpyDeviceToDevice, stream);
}
